// MoELayer_80015240724582
// MI455X (gfx1250) — matrix operand order repaired and checked by reading, not compiled, not run on hardware
//
#include <hip/hip_runtime.h>
#include <hip/hip_bf16.h>

// ---------------- problem constants ----------------
#define BB 4
#define SS 2048
#define DD 512
#define HH 2048
#define EE 8
#define TT (BB * SS)             // 8192 tokens
#define TILE_M 64                // tokens per expert block
#define MAX_TILES (TT / TILE_M)  // 128
#define NCHUNK (HH / 64)         // 32

typedef __attribute__((ext_vector_type(16))) __bf16 v16bf;
typedef __attribute__((ext_vector_type(8)))  float  v8f;
typedef __attribute__((ext_vector_type(4)))  unsigned int u32x4;
typedef __attribute__((ext_vector_type(8)))  int  i32x8;
typedef __attribute__((ext_vector_type(4)))  int  i32x4;

#if defined(__HIP_DEVICE_COMPILE__) && __has_builtin(__builtin_amdgcn_tensor_load_to_lds) && __has_builtin(__builtin_amdgcn_s_wait_tensorcnt)
#define USE_TDM 1
#else
#define USE_TDM 0
#endif

union FragU { u32x4 u[2]; v16bf v; };

// split f32 into bf16 hi + bf16 lo (x ~= hi + lo)
__device__ __forceinline__ void split_bf16(float f, unsigned short& hi, unsigned short& lo) {
    __bf16 h = (__bf16)f;
    __bf16 l = (__bf16)(f - (float)h);
    hi = __builtin_bit_cast(unsigned short, h);
    lo = __builtin_bit_cast(unsigned short, l);
}

__device__ __forceinline__ float gelu_exact(float x) {
    return 0.5f * x * (1.0f + erff(x * 0.70710678118654752f));
}

// A-matrix bf16 16x32 fragment from LDS (row-major [row][K]).
__device__ __forceinline__ v16bf ldsA(const unsigned short* base, int stride,
                                      int row0, int k0, int lane) {
    int m = lane & 15, half = lane >> 4;
    const unsigned short* r = base + (row0 + m) * stride;
    FragU f;
    f.u[0] = *(const u32x4*)(r + k0 + half * 8);
    f.u[1] = *(const u32x4*)(r + k0 + 16 + half * 8);
    return f.v;
}

// B-matrix bf16 32x16 fragment from LDS (stored column-major: [col][K]).
__device__ __forceinline__ v16bf ldsB(const unsigned short* base, int stride,
                                      int col0, int k0, int lane) {
    int n = lane & 15, half = lane >> 4;
    const unsigned short* c = base + (col0 + n) * stride + k0 + half * 8;
    FragU f;
    f.u[0] = *(const u32x4*)(c);
    f.u[1] = *(const u32x4*)(c + 16);
    return f.v;
}

__device__ __forceinline__ v8f wmma_bf16(v16bf a, v16bf b, v8f c) {
    return __builtin_amdgcn_wmma_f32_16x16x32_bf16(false, a, false, b, (short)0, c, false, false);
}

__device__ __forceinline__ v8f mm3(v16bf ah, v16bf al, v16bf bh, v16bf bl, v8f c) {
    c = wmma_bf16(ah, bh, c);
    c = wmma_bf16(ah, bl, c);
    c = wmma_bf16(al, bh, c);
    return c;
}

// ---------------- kernel 1: zero the dense output region ----------------
__global__ __launch_bounds__(256) void k_init(float* __restrict__ out, int n) {
    int i = blockIdx.x * 256 + threadIdx.x;
    if (i < n) out[i] = 0.0f;
}

// ---------------- kernel 2: router (logits, top-2 softmax) ----------------
__global__ __launch_bounds__(256) void k_router(const float* __restrict__ x,
                                                const float* __restrict__ rw,
                                                const float* __restrict__ rb,
                                                float* __restrict__ logits_out,
                                                int2* __restrict__ sel,
                                                float2* __restrict__ selw) {
    int tkn = blockIdx.x * 256 + threadIdx.x;
    if (tkn >= TT) return;
    const float* xr = x + (size_t)tkn * DD;
    float acc[EE];
#pragma unroll
    for (int e = 0; e < EE; ++e) acc[e] = 0.0f;
    for (int d = 0; d < DD; ++d) {
        float xv = xr[d];
#pragma unroll
        for (int e = 0; e < EE; ++e) acc[e] += xv * rw[e * DD + d];
    }
#pragma unroll
    for (int e = 0; e < EE; ++e) {
        acc[e] += rb[e];
        logits_out[(size_t)tkn * EE + e] = acc[e];
    }
    int i0 = 0; float v0 = acc[0];
#pragma unroll
    for (int e = 1; e < EE; ++e) if (acc[e] > v0) { v0 = acc[e]; i0 = e; }
    int i1 = -1; float v1 = -3.4e38f;
#pragma unroll
    for (int e = 0; e < EE; ++e) if (e != i0 && acc[e] > v1) { v1 = acc[e]; i1 = e; }
    float w0 = 1.0f / (1.0f + expf(v1 - v0));
    float w1 = 1.0f - w0;
    sel[tkn]  = make_int2(i0, i1);
    selw[tkn] = make_float2(w0, w1);
}

// ---------------- kernel 3: deterministic per-expert list build ----------------
__global__ __launch_bounds__(256) void k_build(const int2* __restrict__ sel,
                                               const float2* __restrict__ selw,
                                               int* __restrict__ counts,
                                               int* __restrict__ tok_ids,
                                               float* __restrict__ tok_w) {
    int lane = threadIdx.x & 31;
    int e = threadIdx.x >> 5;   // one wave per expert
    int base = 0;
    for (int i0 = 0; i0 < TT; i0 += 32) {
        int tkn = i0 + lane;
        int2 s = sel[tkn];
        float2 w = selw[tkn];
        bool m0 = (s.x == e), m1 = (s.y == e);
        bool m = m0 || m1;
        unsigned mask = __builtin_amdgcn_ballot_w32(m);
        unsigned pre = __builtin_popcount(mask & ((1u << lane) - 1u));
        if (m) {
            int pos = base + (int)pre;
            tok_ids[e * TT + pos] = tkn;
            tok_w[e * TT + pos] = m0 ? w.x : w.y;
        }
        base += __builtin_popcount(mask);
    }
    if (lane == 0) counts[e] = base;
}

// ---------------- kernel 4: fused expert FFN via WMMA + TDM ----------------
// grid = EE * MAX_TILES blocks, 256 threads (8 waves)
// 64-token tile; h chunk = 64; 128-wide panels (4 K-panels GEMM1 + 4 N-panels GEMM2 per chunk)
#define WB_STRIDE1 136   // w1 panel bf16: [64 cols][128 K] + pad  (272B, 16B mult)
#define WB_STRIDE2 72    // w2 panel bf16: [128 cols][64 K] + pad  (144B, 16B mult)
#define HS_STRIDE  72    // h buffer: [64 rows][64 K] + pad
#define WB_ELEMS   9216  // max(64*136=8704, 128*72=9216)
#define PANEL_F32  8192  // 128x64 or 64x128 f32 elements

#if USE_TDM
// Issue one TDM 2-D tile load (f32) into LDS stage buffer. ISA §8.3/8.4 descriptor.
__device__ __forceinline__ void tdm_issue(int hc, int tp, const float* w1e, const float* w2e,
                                          unsigned lds_addr) {
    int hbase = hc * 64;
    unsigned long long ga;
    unsigned d0, d1, t0, t1;
    unsigned long long s0;
    if (tp < 4) {   // GEMM1 K-panel: tile [128 rows of w1][64 cols], row stride HH
        int kp = tp * 128;
        ga = (unsigned long long)(const void*)(w1e + (size_t)kp * HH + hbase);
        d0 = HH; d1 = DD; t0 = 64; t1 = 128; s0 = HH;
    } else {        // GEMM2 N-panel: tile [64 rows of w2][128 cols], row stride DD
        int nbase = (tp - 4) * 128;
        ga = (unsigned long long)(const void*)(w2e + (size_t)hbase * DD + nbase);
        d0 = DD; d1 = HH; t0 = 128; t1 = 64; s0 = DD;
    }
    u32x4 g0;
    g0[0] = 1u;                                  // count=1, user mode, no gather
    g0[1] = lds_addr;                            // LDS byte address
    g0[2] = (unsigned)ga;                        // global_addr[31:0]
    g0[3] = ((unsigned)(ga >> 32) & 0x01FFFFFFu) | (2u << 30);  // addr[56:32], type=2
    i32x8 g1;
    g1[0] = (int)(2u << 16);                     // data_size=2 (4B), mask=0, no pad/iter
    g1[1] = (int)((d0 & 0xFFFFu) << 16);         // tensor_dim0[15:0]
    g1[2] = (int)(((d0 >> 16) & 0xFFFFu) | ((d1 & 0xFFFFu) << 16));
    g1[3] = (int)(((d1 >> 16) & 0xFFFFu) | (t0 << 16));   // tile_dim0
    g1[4] = (int)t1;                             // tile_dim1 (tile_dim2=0)
    g1[5] = (int)(unsigned)(s0 & 0xFFFFFFFFu);   // tensor_dim0_stride[31:0]
    g1[6] = (int)(unsigned)((s0 >> 32) & 0xFFFFu);
    g1[7] = 0;                                   // tensor_dim1_stride = 0 (2-D)
    i32x4 gz4 = (i32x4)0;
    i32x8 gz8 = (i32x8)0;
    __builtin_amdgcn_tensor_load_to_lds(g0, g1, gz4, gz4, gz8, 0);
}
#endif

__global__ void __launch_bounds__(256, 2) k_expert(const float* __restrict__ x,
                                                   const float* __restrict__ w1,
                                                   const float* __restrict__ b1,
                                                   const float* __restrict__ w2,
                                                   const float* __restrict__ b2,
                                                   const int* __restrict__ counts,
                                                   const int* __restrict__ tok_ids,
                                                   const float* __restrict__ tok_w,
                                                   float* __restrict__ out) {
    const int e = blockIdx.x / MAX_TILES;
    const int tile = blockIdx.x % MAX_TILES;
    const int n = counts[e];
    if (tile * TILE_M >= n) return;

    __shared__ unsigned short xs_hi[TILE_M * DD];        // 64KB
    __shared__ unsigned short xs_lo[TILE_M * DD];        // 64KB
    __shared__ float          stagef[2][PANEL_F32];      // 64KB (TDM double buffer)
    __shared__ unsigned short wb_hi[WB_ELEMS];           // 18KB
    __shared__ unsigned short wb_lo[WB_ELEMS];           // 18KB
    __shared__ unsigned short hs_hi[TILE_M * HS_STRIDE]; // 9KB
    __shared__ unsigned short hs_lo[TILE_M * HS_STRIDE]; // 9KB
    __shared__ int   s_tok[TILE_M];
    __shared__ float s_wgt[TILE_M];

    const int t = threadIdx.x;
    const int lane = t & 31;
    const int wid = t >> 5;

    if (t < TILE_M) {
        int i = tile * TILE_M + t;
        int id = 0; float w = 0.0f;
        if (i < n) { id = tok_ids[e * TT + i]; w = tok_w[e * TT + i]; }
        s_tok[t] = id; s_wgt[t] = w;
    }
    __syncthreads();

    // stage x tile [64 x 512] as bf16 hi/lo
    for (int idx = t; idx < TILE_M * DD; idx += 256) {
        int row = idx >> 9;
        int col = idx & 511;
        float f = x[(size_t)s_tok[row] * DD + col];
        split_bf16(f, xs_hi[idx], xs_lo[idx]);
    }

    const float* w1e = w1 + (size_t)e * DD * HH;
    const float* w2e = w2 + (size_t)e * HH * DD;
    const float* b1e = b1 + (size_t)e * HH;
    const float* b2e = b2 + (size_t)e * DD;

    const int rt1 = wid & 3;          // GEMM1 row tile, wave owns col tiles cb1..cb1+1
    const int cb1 = (wid >> 2) * 2;
    const int rt2 = wid & 3;          // GEMM2 row tile, wave owns 4 col-tiles/panel
    const int cg2 = wid >> 2;

    v8f accy[16];
#pragma unroll
    for (int i = 0; i < 16; ++i) accy[i] = (v8f)0.0f;

    const int half = lane >> 4, nl = lane & 15;

#if USE_TDM
    unsigned stage_addr[2];
    stage_addr[0] = (unsigned)(size_t)(void*)&stagef[0][0];
    stage_addr[1] = (unsigned)(size_t)(void*)&stagef[1][0];
    if (wid == 0) tdm_issue(0, 0, w1e, w2e, stage_addr[0]);   // prologue: panel (0,0)
#endif

    for (int hc = 0; hc < NCHUNK; ++hc) {
        const int hbase = hc * 64;
        v8f a0[2], a1[2], a2[2];
#pragma unroll
        for (int j = 0; j < 2; ++j) { a0[j] = (v8f)0.0f; a1[j] = (v8f)0.0f; a2[j] = (v8f)0.0f; }

        // ---- GEMM1: 4 K-panels of 128 (phases 0..3, stage buf = p&1)
#pragma unroll
        for (int p = 0; p < 4; ++p) {
            const int buf = p & 1;
#if USE_TDM
            if (wid == 0) __builtin_amdgcn_s_wait_tensorcnt(0);   // panel (hc,p) landed
#endif
            __syncthreads();   // stage[buf] visible; wbuf free
#if USE_TDM
            if (wid == 0)      // prefetch next phase (p+1 <= 4, same chunk)
                tdm_issue(hc, p + 1, w1e, w2e, stage_addr[(p + 1) & 1]);
#endif
            {   // convert: f32 -> bf16 hi/lo transposed into wbuf
                int col = t & 63, kg = t >> 6;
#pragma unroll 4
                for (int j = 0; j < 32; ++j) {
                    int k = kg * 32 + j;
#if USE_TDM
                    float f = stagef[buf][k * 64 + col];
#else
                    float f = w1e[(size_t)(p * 128 + k) * HH + hbase + col];
#endif
                    split_bf16(f, wb_hi[col * WB_STRIDE1 + k], wb_lo[col * WB_STRIDE1 + k]);
                }
            }
            __syncthreads();
            const int kp = p * 128;
#pragma unroll
            for (int k0 = 0; k0 < 128; k0 += 32) {
                v16bf xh = ldsA(xs_hi, DD, rt1 * 16, kp + k0, lane);
                v16bf xl = ldsA(xs_lo, DD, rt1 * 16, kp + k0, lane);
#pragma unroll
                for (int j = 0; j < 2; ++j) {
                    v16bf bh = ldsB(wb_hi, WB_STRIDE1, (cb1 + j) * 16, k0, lane);
                    v16bf bl = ldsB(wb_lo, WB_STRIDE1, (cb1 + j) * 16, k0, lane);
                    a0[j] = wmma_bf16(xh, bh, a0[j]);
                    a1[j] = wmma_bf16(xh, bl, a1[j]);
                    a2[j] = wmma_bf16(xl, bh, a2[j]);
                }
            }
        }

        // ---- bias + exact GELU -> hs (consumed after the next barrier)
#pragma unroll
        for (int j = 0; j < 2; ++j) {
            int colL = (cb1 + j) * 16 + nl;
            float bias = b1e[hbase + colL];
#pragma unroll
            for (int g = 0; g < 8; ++g) {
                int rowL = rt1 * 16 + half * 8 + g;
                float v = gelu_exact(a0[j][g] + a1[j][g] + a2[j][g] + bias);
                split_bf16(v, hs_hi[rowL * HS_STRIDE + colL], hs_lo[rowL * HS_STRIDE + colL]);
            }
        }

        // ---- GEMM2: 4 N-panels of 128 (phases 4..7, stage buf = (4+q)&1)
#pragma unroll
        for (int q = 0; q < 4; ++q) {
            const int buf = (4 + q) & 1;
#if USE_TDM
            if (wid == 0) __builtin_amdgcn_s_wait_tensorcnt(0);   // panel (hc,4+q) landed
#endif
            __syncthreads();
#if USE_TDM
            if (wid == 0 && !(hc == NCHUNK - 1 && q == 3)) {
                int hcn = (q == 3) ? hc + 1 : hc;
                int tpn = (q == 3) ? 0 : 5 + q;
                tdm_issue(hcn, tpn, w1e, w2e, stage_addr[(5 + q) & 1]);
            }
#endif
            {   // convert
                int col = t & 127, kg = t >> 7;
#pragma unroll 4
                for (int j = 0; j < 32; ++j) {
                    int k = kg * 32 + j;
#if USE_TDM
                    float f = stagef[buf][k * 128 + col];
#else
                    float f = w2e[(size_t)(hbase + k) * DD + q * 128 + col];
#endif
                    split_bf16(f, wb_hi[col * WB_STRIDE2 + k], wb_lo[col * WB_STRIDE2 + k]);
                }
            }
            __syncthreads();
#pragma unroll
            for (int k0 = 0; k0 < 64; k0 += 32) {
                v16bf hh = ldsA(hs_hi, HS_STRIDE, rt2 * 16, k0, lane);
                v16bf hl = ldsA(hs_lo, HS_STRIDE, rt2 * 16, k0, lane);
#pragma unroll
                for (int j = 0; j < 4; ++j) {
                    int ct = cg2 * 4 + j;
                    v16bf bh = ldsB(wb_hi, WB_STRIDE2, ct * 16, k0, lane);
                    v16bf bl = ldsB(wb_lo, WB_STRIDE2, ct * 16, k0, lane);
                    accy[q * 4 + j] = mm3(hh, hl, bh, bl, accy[q * 4 + j]);
                }
            }
        }
    }

    // ---- epilogue: + b2, scale by routing weight, scatter-add into out
#pragma unroll
    for (int q4 = 0; q4 < 4; ++q4)
#pragma unroll
        for (int j = 0; j < 4; ++j) {
            int col = q4 * 128 + (cg2 * 4 + j) * 16 + nl;
            float bias = b2e[col];
#pragma unroll
            for (int g = 0; g < 8; ++g) {
                int rowL = rt2 * 16 + half * 8 + g;
                float v = (accy[q4 * 4 + j][g] + bias) * s_wgt[rowL];
                unsafeAtomicAdd(&out[(size_t)s_tok[rowL] * DD + col], v);
            }
        }
}

// ---------------- host launcher ----------------
extern "C" void kernel_launch(void* const* d_in, const int* in_sizes, int n_in,
                              void* d_out, int out_size, void* d_ws, size_t ws_size,
                              hipStream_t stream) {
    const float* x        = (const float*)d_in[0];
    const float* router_w = (const float*)d_in[1];
    const float* router_b = (const float*)d_in[2];
    const float* w1       = (const float*)d_in[3];
    const float* b1       = (const float*)d_in[4];
    const float* w2       = (const float*)d_in[5];
    const float* b2       = (const float*)d_in[6];
    // d_in[7] = top_k (== 2, compiled-in)

    float* out    = (float*)d_out;
    float* logits = out + (size_t)TT * DD;   // second tuple output

    char* ws = (char*)d_ws;
    int*    counts  = (int*)ws;                                  // 8 ints
    int2*   sel     = (int2*)(ws + 256);                         // T*8B
    float2* selw    = (float2*)(ws + 256 + (size_t)TT * 8);      // T*8B
    int*    tok_ids = (int*)(ws + 256 + (size_t)TT * 16);        // E*T*4B
    float*  tok_w   = (float*)(ws + 256 + (size_t)TT * 16 + (size_t)EE * TT * 4);

    k_init<<<(TT * DD + 255) / 256, 256, 0, stream>>>(out, TT * DD);
    k_router<<<TT / 256, 256, 0, stream>>>(x, router_w, router_b, logits, sel, selw);
    k_build<<<1, 256, 0, stream>>>(sel, selw, counts, tok_ids, tok_w);
    k_expert<<<EE * MAX_TILES, 256, 0, stream>>>(x, w1, b1, w2, b2,
                                                 counts, tok_ids, tok_w, out);
}
